// GeneratorBlock_86552180949509
// MI455X (gfx1250) — matrix operand order repaired and checked by reading, not compiled, not run on hardware
//
#include <hip/hip_runtime.h>

// ---------------------------------------------------------------------------
// StyleGAN2 generator block on gfx1250 (MI455X).
// 3x3 modulated/demodulated conv as 9-tap-decomposed implicit GEMM using
// v_wmma_f32_16x16x32_bf16 (wave32). Style/(1/sqrt(fan_in)) folded into bf16
// inputs; demod+bias+lrelu (+next-layer modulation) fused into GEMM epilogue.
// B operand staged once per cin-chunk as a 4x64 halo tile in LDS and reused
// by all 9 taps via shifted per-lane addressing (zero-row for padding).
// ---------------------------------------------------------------------------

typedef __attribute__((ext_vector_type(16))) __bf16 v16bf;
typedef __attribute__((ext_vector_type(8)))  float  v8f;

union FragU { v16bf v; uint4 q[2]; };

#define DEVI __device__ __forceinline__

DEVI unsigned short f2bf(float f) {
  unsigned int u = __float_as_uint(f);
  u += 0x7fffu + ((u >> 16) & 1u);          // round-to-nearest-even
  return (unsigned short)(u >> 16);
}

// sizes
static constexpr int NB   = 16;
static constexpr int NC   = 512;
static constexpr int HW   = 4096;   // 64x64
static constexpr float C_AFF   = 0.044194173824159216f;  // 1/sqrt(512)
static constexpr float C_CONV  = 0.014731391274719739f;  // 1/sqrt(512*9)
static constexpr float CC_CONV = 1.0f / 4608.0f;         // C_CONV^2
static constexpr float C_RGB   = 0.044194173824159216f;  // 1/sqrt(512)

// ---------------------------------------------------------------------------
// s[b,i] = (1/sqrt(512)) * sum_d w[b,d]*A[i,d] + bias[i]
// 16 adjacent lanes share one A row -> broadcast reads.
// ---------------------------------------------------------------------------
__global__ __launch_bounds__(256) void affine_kernel(
    const float* __restrict__ w, const float* __restrict__ A,
    const float* __restrict__ bias, float* __restrict__ s) {
  int idx = blockIdx.x * 256 + threadIdx.x;     // 0..8191
  int b = idx & 15, i = idx >> 4;
  const float* wr = w + b * NC;
  const float* ar = A + i * NC;
  float sum = 0.f;
  for (int d = 0; d < NC; ++d) sum += wr[d] * ar[d];
  s[b * NC + i] = sum * C_AFF + bias[i];
}

// ---------------------------------------------------------------------------
// conv weight (Cout,Cin,3,3) f32 -> wb[tap][cout][cin] bf16; q[cout,cin]=sum w^2
// ---------------------------------------------------------------------------
__global__ __launch_bounds__(256) void wq_kernel(
    const float* __restrict__ cw, unsigned short* __restrict__ wb,
    float* __restrict__ q) {
  int idx = blockIdx.x * 256 + threadIdx.x;     // 0..262143
  int cout = idx >> 9, cin = idx & 511;
  const float* src = cw + idx * 9;
  float sum = 0.f;
#pragma unroll
  for (int tap = 0; tap < 9; ++tap) {
    float f = src[tap];
    sum += f * f;
    wb[(tap * NC + cout) * NC + cin] = f2bf(f);
  }
  q[idx] = sum;
}

// ---------------------------------------------------------------------------
// sigma_inv[b,cout] = rsqrt(c^2 * sum_cin q[cout,cin]*s[b,cin]^2 + 1e-8)
// 16 adjacent lanes share one q row -> broadcast reads.
// ---------------------------------------------------------------------------
__global__ __launch_bounds__(256) void sigma_kernel(
    const float* __restrict__ q, const float* __restrict__ s,
    float* __restrict__ sig) {
  int idx = blockIdx.x * 256 + threadIdx.x;     // 0..8191
  int b = idx & 15, cout = idx >> 4;
  const float* qr = q + cout * NC;
  const float* sr = s + b * NC;
  float sum = 0.f;
  for (int c = 0; c < NC; ++c) { float sv = sr[c]; sum += qr[c] * sv * sv; }
  sig[b * NC + cout] = rsqrtf(sum * CC_CONV + 1e-8f);
}

// ---------------------------------------------------------------------------
// xs[b,c,p] = bf16( x[b,c,p] * s[b,c] * C_CONV )   (4 elems / thread)
// ---------------------------------------------------------------------------
__global__ __launch_bounds__(256) void premod_kernel(
    const float* __restrict__ x, const float* __restrict__ s,
    unsigned short* __restrict__ xs) {
  int idx = blockIdx.x * 256 + threadIdx.x;
  int e = idx * 4;
  int bc = e >> 12;                              // (b*512+c), HW=4096
  float sc = s[bc] * C_CONV;
  const float4 xv = *(const float4*)(x + e);
  unsigned int lo = (unsigned int)f2bf(xv.x * sc) |
                    ((unsigned int)f2bf(xv.y * sc) << 16);
  unsigned int hi = (unsigned int)f2bf(xv.z * sc) |
                    ((unsigned int)f2bf(xv.w * sc) << 16);
  uint2 t; t.x = lo; t.y = hi;
  *(uint2*)(xs + e) = t;
}

// ---------------------------------------------------------------------------
// 3x3 modulated conv, 9-tap implicit GEMM with WMMA bf16.
// Block: 256 thr (8 waves), tile 128(M=cout) x 128(N=spatial), K-step 32.
// Per kc: one 4x64 halo X-tile (LDS, [p][cin]) serves all 9 taps' B operands.
// MODE 0: out -> bf16 next-layer premod input (lrelu(acc*sig+b)*s2*c2)
// MODE 1: out -> f32 h output (lrelu(acc*sig+b))
// ---------------------------------------------------------------------------
template <int MODE>
__global__ __launch_bounds__(256) void modconv3x3_wmma(
    const unsigned short* __restrict__ wtap,   // [9][512][512] bf16
    const unsigned short* __restrict__ xs,     // [B][512][4096] bf16 (premod)
    const float* __restrict__ sigma,           // [B][512]
    const float* __restrict__ bias,            // [512]
    const float* __restrict__ snext,           // [B][512] (MODE 0)
    float cnext,
    unsigned short* __restrict__ obf,          // MODE 0 out
    float* __restrict__ of32) {                // MODE 1 out
  __shared__ __align__(16) unsigned short Als[128 * 40];  // A: [cout][k] pad 40
  __shared__ __align__(16) unsigned short Xls[257 * 40];  // X: [4x64 p][cin] pad 40; row 256 = zeros

  const int tid  = threadIdx.x;
  const int lane = tid & 31;
  const int wave = tid >> 5;
  const int mwave = wave & 1, nwave = wave >> 1;   // 2x4 wave grid
  const int b = blockIdx.z, mtile = blockIdx.y, ntile = blockIdx.x;

  // zero padding row (read by out-of-image lanes)
  if (tid < 40) Xls[256 * 40 + tid] = 0;

  v8f acc[4][2];
#pragma unroll
  for (int i = 0; i < 4; ++i)
#pragma unroll
    for (int j = 0; j < 2; ++j)
#pragma unroll
      for (int e = 0; e < 8; ++e) acc[i][j][e] = 0.0f;

  // A-tile fill indices (each thread: 16 bf16 of one cout row)
  const int arow = tid & 127;
  const int akh  = (tid >> 7) << 4;               // 0 or 16
  const unsigned short* abase = wtap + (mtile * 128 + arow) * NC + akh;

  // X halo-tile fill indices: window rows ntile*2-1 .. ntile*2+2, 64 cols
  const int fxc  = tid & 63;                      // gx
  const int frow = tid >> 6;                      // 0..3
  const int wy0  = ntile * 2 - 1;
  const int gy   = wy0 + frow;
  const bool gvalid = ((unsigned)gy < 64u);
  const unsigned short* xrow = xs + (size_t)b * NC * HW + gy * 64 + fxc;
  unsigned short* xdst = &Xls[tid * 40];

  // fragment lane addressing (contiguous k = 16*(lane/16) + s for both A and B)
  const int fr  = lane & 15;
  const int kbA = (lane & 16) ? 16 : 0;           // A: K {0..15} vs {16..31}
  const int kbB = (lane & 16) ? 16 : 0;           // B: K {0..15} vs {16..31}
  const int pg0 = ntile * 128 + nwave * 32 + fr;  // spatial pos, ni=0
  const int pg1 = pg0 + 16;                       // spatial pos, ni=1
  const int py0 = pg0 >> 6, px0 = pg0 & 63;
  const int py1 = pg1 >> 6, px1 = pg1 & 63;

  for (int kc = 0; kc < 16; ++kc) {
    for (int tap = 0; tap < 9; ++tap) {
      __syncthreads();
      if (tap == 0 && gvalid) {                   // halo X-tile, once per kc
        const unsigned short* src = xrow + kc * 32 * HW;
#pragma unroll
        for (int g = 0; g < 4; ++g) {
          union { unsigned short h[8]; uint4 q; } pk;
#pragma unroll
          for (int j = 0; j < 8; ++j) pk.h[j] = src[(g * 8 + j) * HW];
          *(uint4*)&xdst[g * 8] = pk.q;
        }
      }
      {                                           // A tile [128 cout][32 cin]
        const unsigned short* srcA = abase + tap * NC * NC + kc * 32;
        uint4 d0 = *(const uint4*)(srcA);
        uint4 d1 = *(const uint4*)(srcA + 8);
        *(uint4*)&Als[arow * 40 + akh]     = d0;
        *(uint4*)&Als[arow * 40 + akh + 8] = d1;
        __builtin_prefetch(srcA + NC * NC, 0, 0); // next tap's A (global_prefetch)
      }
      __syncthreads();

      const int rr = tap / 3 - 1, tt = tap % 3 - 1;
      const int sy0 = py0 + rr, sx0 = px0 + tt;
      const int sy1 = py1 + rr, sx1 = px1 + tt;
      const bool v0 = ((unsigned)sy0 < 64u) && ((unsigned)sx0 < 64u);
      const bool v1 = ((unsigned)sy1 < 64u) && ((unsigned)sx1 < 64u);
      const int l0 = v0 ? ((sy0 - wy0) * 64 + sx0) : 256;
      const int l1 = v1 ? ((sy1 - wy0) * 64 + sx1) : 256;

      FragU bf0, bf1;
      bf0.q[0] = *(const uint4*)&Xls[l0 * 40 + kbB];
      bf0.q[1] = *(const uint4*)&Xls[l0 * 40 + kbB + 8];
      bf1.q[0] = *(const uint4*)&Xls[l1 * 40 + kbB];
      bf1.q[1] = *(const uint4*)&Xls[l1 * 40 + kbB + 8];
#pragma unroll
      for (int mi = 0; mi < 4; ++mi) {
        FragU af;
        const int row = mwave * 64 + mi * 16 + fr;
        af.q[0] = *(const uint4*)&Als[row * 40 + kbA];
        af.q[1] = *(const uint4*)&Als[row * 40 + kbA + 8];
        acc[mi][0] = __builtin_amdgcn_wmma_f32_16x16x32_bf16(
            false, af.v, false, bf0.v, (short)0, acc[mi][0], false, false);
        acc[mi][1] = __builtin_amdgcn_wmma_f32_16x16x32_bf16(
            false, af.v, false, bf1.v, (short)0, acc[mi][1], false, false);
      }
    }
  }

  // epilogue: demod + bias + leaky-relu (+ next-layer modulation for MODE 0)
  const int laneHi8 = (lane & 16) ? 8 : 0;
#pragma unroll
  for (int mi = 0; mi < 4; ++mi) {
#pragma unroll
    for (int ni = 0; ni < 2; ++ni) {
      const int p = (ni == 0) ? pg0 : pg1;
#pragma unroll
      for (int e = 0; e < 8; ++e) {
        const int cout = mtile * 128 + mwave * 64 + mi * 16 + e + laneHi8;
        float v = acc[mi][ni][e];
        v = v * sigma[b * NC + cout] + bias[cout];
        v = fmaxf(v, 0.2f * v);                  // leaky relu 0.2
        if (MODE == 0) {
          obf[(b * NC + cout) * HW + p] = f2bf(v * snext[b * NC + cout] * cnext);
        } else {
          of32[(b * NC + cout) * HW + p] = v;
        }
      }
    }
  }
}

// ---------------------------------------------------------------------------
// ToRGB: rgb[b,j,p] = sum_c rgb_w[j,c]*c3*s3[b,c]*h[b,c,p] + rgb_b[j]
// ---------------------------------------------------------------------------
__global__ __launch_bounds__(256) void torgb_kernel(
    const float* __restrict__ h, const float* __restrict__ s3,
    const float* __restrict__ rgbw, const float* __restrict__ rgbb,
    float* __restrict__ out) {
  __shared__ float coef[3][512];
  const int b = blockIdx.y;
  const int tid = threadIdx.x;
  for (int i = tid; i < 3 * NC; i += 256) {
    int j = i >> 9, c = i & 511;
    coef[j][c] = rgbw[j * NC + c] * C_RGB * s3[b * NC + c];
  }
  __syncthreads();
  const int p = blockIdx.x * 256 + tid;
  float a0 = 0.f, a1 = 0.f, a2 = 0.f;
  const float* hb = h + b * NC * HW + p;
  for (int c = 0; c < NC; ++c) {
    float hv = hb[c * HW];
    a0 += coef[0][c] * hv;
    a1 += coef[1][c] * hv;
    a2 += coef[2][c] * hv;
  }
  out[(b * 3 + 0) * HW + p] = a0 + rgbb[0];
  out[(b * 3 + 1) * HW + p] = a1 + rgbb[1];
  out[(b * 3 + 2) * HW + p] = a2 + rgbb[2];
}

// ---------------------------------------------------------------------------
extern "C" void kernel_launch(void* const* d_in, const int* in_sizes, int n_in,
                              void* d_out, int out_size, void* d_ws, size_t ws_size,
                              hipStream_t stream) {
  const float* x       = (const float*)d_in[0];
  const float* w       = (const float*)d_in[1];
  const float* aff1_w  = (const float*)d_in[2];
  const float* aff1_b  = (const float*)d_in[3];
  const float* conv1_w = (const float*)d_in[4];
  const float* conv1_b = (const float*)d_in[5];
  const float* aff2_w  = (const float*)d_in[6];
  const float* aff2_b  = (const float*)d_in[7];
  const float* conv2_w = (const float*)d_in[8];
  const float* conv2_b = (const float*)d_in[9];
  const float* rgb_aff_w = (const float*)d_in[10];
  const float* rgb_aff_b = (const float*)d_in[11];
  const float* rgb_w   = (const float*)d_in[12];
  const float* rgb_b   = (const float*)d_in[13];

  // workspace carve-up (all sizes 256B aligned)
  char* ws = (char*)d_ws;
  size_t off = 0;
  const size_t SZ_S  = (size_t)NB * NC * sizeof(float);               // 32 KB
  const size_t SZ_Q  = (size_t)NC * NC * sizeof(float);               // 1 MB
  const size_t SZ_WB = (size_t)9 * NC * NC * sizeof(unsigned short);  // 4.5 MB
  const size_t SZ_XS = (size_t)NB * NC * HW * sizeof(unsigned short); // 64 MB
  float* s1   = (float*)(ws + off); off += SZ_S;
  float* s2   = (float*)(ws + off); off += SZ_S;
  float* s3   = (float*)(ws + off); off += SZ_S;
  float* sig1 = (float*)(ws + off); off += SZ_S;
  float* sig2 = (float*)(ws + off); off += SZ_S;
  float* q1   = (float*)(ws + off); off += SZ_Q;
  float* q2   = (float*)(ws + off); off += SZ_Q;
  unsigned short* w1b = (unsigned short*)(ws + off); off += SZ_WB;
  unsigned short* w2b = (unsigned short*)(ws + off); off += SZ_WB;
  unsigned short* xs1 = (unsigned short*)(ws + off); off += SZ_XS;
  unsigned short* xs2 = (unsigned short*)(ws + off); off += SZ_XS;

  float* hout   = (float*)d_out;                 // (16,512,64,64)
  float* rgbout = hout + (size_t)NB * NC * HW;   // (16,3,64,64)

  // 1) styles
  affine_kernel<<<32, 256, 0, stream>>>(w, aff1_w, aff1_b, s1);
  affine_kernel<<<32, 256, 0, stream>>>(w, aff2_w, aff2_b, s2);
  affine_kernel<<<32, 256, 0, stream>>>(w, rgb_aff_w, rgb_aff_b, s3);

  // 2) weight convert + per-(cout,cin) squared sums
  wq_kernel<<<1024, 256, 0, stream>>>(conv1_w, w1b, q1);
  wq_kernel<<<1024, 256, 0, stream>>>(conv2_w, w2b, q2);

  // 3) demod factors
  sigma_kernel<<<32, 256, 0, stream>>>(q1, s1, sig1);
  sigma_kernel<<<32, 256, 0, stream>>>(q2, s2, sig2);

  // 4) pre-modulate conv1 input to bf16
  premod_kernel<<<32768, 256, 0, stream>>>(x, s1, xs1);

  // 5) conv1 (WMMA) -> bf16 pre-modulated conv2 input
  dim3 ggrid(HW / 128, NC / 128, NB);   // (32, 4, 16)
  modconv3x3_wmma<0><<<ggrid, 256, 0, stream>>>(
      w1b, xs1, sig1, conv1_b, s2, C_CONV, xs2, nullptr);

  // 6) conv2 (WMMA) -> f32 h output
  modconv3x3_wmma<1><<<ggrid, 256, 0, stream>>>(
      w2b, xs2, sig2, conv2_b, nullptr, 0.f, nullptr, hout);

  // 7) ToRGB
  dim3 rgrid(HW / 256, NB);             // (16, 16)
  torgb_kernel<<<rgrid, 256, 0, stream>>>(hout, s3, rgb_w, rgb_b, rgbout);

  (void)in_sizes; (void)n_in; (void)out_size; (void)ws_size;
}
